// Elastic_Matching_Model_14027363188926
// MI455X (gfx1250) — compile-verified, matrix operand order checked by reading, not run on hardware
//
#include <hip/hip_runtime.h>
#include <cmath>

// ---------------------------------------------------------------------------
// MI455X (gfx1250) implementation of the Elastic Matching model.
//
// Key algorithmic move: conv3x3 -> global-avg-pool is commuted into
// 9-shifted-window sums (one streaming pass over x1/x2, ~103MB => the HBM
// roofline floor at 23.3 TB/s) followed by a 128x128xK=18432 GEMM.
// All GEMMs use v_wmma_f32_16x16x32_bf16 (wave32, 16x16 tiles).
// ---------------------------------------------------------------------------

#define B_   8
#define T_   16
#define DIM_ 128
#define CIN_ 2048
#define NSTEP_ 14
#define FIX_ 3
#define DFF_ 2048

typedef __attribute__((ext_vector_type(16))) __bf16         v16bf;
typedef __attribute__((ext_vector_type(8)))  float          v8f;
typedef __attribute__((ext_vector_type(8)))  unsigned short u16x8;
struct U256 { u16x8 lo, hi; };

__device__ __forceinline__ unsigned short f2bf(float f) {
  unsigned int u = __builtin_bit_cast(unsigned int, f);
  u += 0x7FFFu + ((u >> 16) & 1u);              // round-to-nearest-even
  return (unsigned short)(u >> 16);
}

// -------------------------- generic f32 -> bf16 ----------------------------
__global__ __launch_bounds__(256) void k_f2bf(const float* __restrict__ in,
                                              unsigned short* __restrict__ out, int n) {
  int i = blockIdx.x * blockDim.x + threadIdx.x;
  if (i < n) out[i] = f2bf(in[i]);
}

// ---------------- conv3x3+GAP commuted: 9 shifted-window sums --------------
// x: (N*C, 49) f32 -> S: (N*C, 9) bf16, S[ky*3+kx] = sum over clipped window.
__global__ __launch_bounds__(256) void k_reduce_gap(const float* __restrict__ x,
                                                    unsigned short* __restrict__ S) {
  int p = blockIdx.x * blockDim.x + threadIdx.x;       // 128*2048 threads
  const float* px = x + (size_t)p * 49;
  float c0 = 0.f, c1 = 0.f, c2 = 0.f;                  // col subsets over all rows
  float r00 = 0.f, r01 = 0.f, r02 = 0.f;               // row 0 contributions
  float r60 = 0.f, r61 = 0.f, r62 = 0.f;               // row 6 contributions
  #pragma unroll
  for (int y = 0; y < 7; ++y) {
    float v0 = px[y*7+0], v1 = px[y*7+1], v2 = px[y*7+2], v3 = px[y*7+3];
    float v4 = px[y*7+4], v5 = px[y*7+5], v6 = px[y*7+6];
    float full = v0+v1+v2+v3+v4+v5+v6;
    float left = full - v6, right = full - v0;         // cols 0..5 / 1..6
    c0 += left; c1 += full; c2 += right;
    if (y == 0) { r00 = left; r01 = full; r02 = right; }
    if (y == 6) { r60 = left; r61 = full; r62 = right; }
  }
  unsigned short* o = S + (size_t)p * 9;
  o[0] = f2bf(c0 - r60); o[1] = f2bf(c1 - r61); o[2] = f2bf(c2 - r62);   // ky=0: rows 0..5
  o[3] = f2bf(c0);       o[4] = f2bf(c1);       o[5] = f2bf(c2);         // ky=1: rows 0..6
  o[6] = f2bf(c0 - r00); o[7] = f2bf(c1 - r01); o[8] = f2bf(c2 - r02);   // ky=2: rows 1..6
}

// --------------------------- WMMA bf16 GEMM (NT) ---------------------------
// C[M,N] = act(scale * A[M,K] x Bt[N,K]^T + bias), one wave per 16x16 tile.
// A/Bt fragments loaded per CDNA5 16-bit layout: lane<16 -> K [0..7]+[16..23],
// lane>=16 -> K [8..15]+[24..31] of its row. K must be a multiple of 32.
__global__ __launch_bounds__(256) void k_gemm_bf16(
    const unsigned short* __restrict__ A, const unsigned short* __restrict__ Bt,
    float* __restrict__ C, unsigned short* __restrict__ Cbf,
    const float* __restrict__ bias, int M, int N, int K, float scale, int relu) {
  const int tilesN = (N + 15) >> 4;
  const int tilesM = (M + 15) >> 4;
  const int wid = blockIdx.x * (blockDim.x >> 5) + (threadIdx.x >> 5);
  if (wid >= tilesM * tilesN) return;                  // wave-uniform exit
  const int tm = wid / tilesN, tn = wid - tm * tilesN;
  const int lane = threadIdx.x & 31;
  const int half = lane >> 4;                          // 0 or 1
  const int l16  = lane & 15;
  int r = tm * 16 + l16; if (r > M - 1) r = M - 1;     // clamp; rows>=M never stored
  int c = tn * 16 + l16; if (c > N - 1) c = N - 1;
  const unsigned short* pa = A  + (size_t)r * K + (half << 3);
  const unsigned short* pb = Bt + (size_t)c * K + (half << 3);
  v8f acc = {};
  for (int k0 = 0; k0 < K; k0 += 32) {
    U256 ab, bb;
    ab.lo = *(const u16x8*)(pa + k0);
    ab.hi = *(const u16x8*)(pa + k0 + 16);
    bb.lo = *(const u16x8*)(pb + k0);
    bb.hi = *(const u16x8*)(pb + k0 + 16);
    v16bf av = __builtin_bit_cast(v16bf, ab);
    v16bf bv = __builtin_bit_cast(v16bf, bb);
    acc = __builtin_amdgcn_wmma_f32_16x16x32_bf16(false, av, false, bv,
                                                  (short)0, acc, false, false);
  }
  const int col = tn * 16 + l16;
  #pragma unroll
  for (int v = 0; v < 8; ++v) {
    const int row = tm * 16 + (half << 3) + v;         // C/D layout: M = v + 8*half
    if (row < M && col < N) {
      float x = acc[v] * scale + (bias ? bias[col] : 0.f);
      if (relu) x = fmaxf(x, 0.f);
      C[(size_t)row * N + col] = x;
      if (Cbf) Cbf[(size_t)row * N + col] = f2bf(x);
    }
  }
}

// ------------- im2col (kernel=3, SAME over T) for the step encoder ---------
__global__ __launch_bounds__(256) void k_im2col(const float* __restrict__ tok,
                                                unsigned short* __restrict__ A) {
  int idx = blockIdx.x * blockDim.x + threadIdx.x;     // 128*384
  if (idx >= (B_*T_) * (DIM_*3)) return;
  int row = idx / (DIM_*3), kk = idx - row * (DIM_*3);
  int cch = kk / 3, j = kk - cch * 3;
  int n = row >> 4, t = row & 15, tt = t + j - 1;
  float v = (tt >= 0 && tt < T_) ? tok[(size_t)(n * T_ + tt) * DIM_ + cch] : 0.f;
  A[idx] = f2bf(v);
}

// -------- cosine similarity + 2D integral image, one block per batch -------
__global__ __launch_bounds__(256) void k_sim_pred(const float* __restrict__ f1,
                                                  const float* __restrict__ f2,
                                                  float* __restrict__ pred) {
  int b = blockIdx.x, t = threadIdx.x, i = t >> 4, j = t & 15;
  __shared__ float sim[256], n1[16], n2[16];
  const float* F1 = f1 + (size_t)b * T_ * DIM_;
  const float* F2 = f2 + (size_t)b * T_ * DIM_;
  float d = 0.f;
  for (int cc = 0; cc < DIM_; ++cc) d += F1[i*DIM_+cc] * F2[j*DIM_+cc];
  if (j == 0) { float s = 0.f; for (int cc = 0; cc < DIM_; ++cc) { float v = F1[i*DIM_+cc]; s += v*v; } n1[i] = sqrtf(s); }
  if (i == 0) { float s = 0.f; for (int cc = 0; cc < DIM_; ++cc) { float v = F2[j*DIM_+cc]; s += v*v; } n2[j] = sqrtf(s); }
  __syncthreads();
  sim[t] = d / fmaxf(n1[i] * n2[j], 1e-8f);
  __syncthreads();
  float acc = 0.f;
  for (int a = 0; a <= i; ++a)
    for (int cc = 0; cc <= j; ++cc) acc += sim[a*16+cc];
  pred[b*256 + t] = acc;
}

// ------------------------------- DP + argmax -------------------------------
__global__ __launch_bounds__(256) void k_dp(const float* __restrict__ pred,
                                            int* __restrict__ inds) {
  int b = blockIdx.x, t = threadIdx.x, i = t >> 4, j = t & 15;
  __shared__ float P[256], D[256];
  P[t] = pred[b*256 + t];
  __syncthreads();
  D[t] = P[t] / (float)((i+1) * (j+1));
  __syncthreads();
  for (int k = 1; k <= NSTEP_ - 1; ++k) {
    float best = -INFINITY; int bi = 0;
    float pij = P[i*16+j];
    for (int a = 0; a < i; ++a)
      for (int cc = 0; cc < j; ++cc) {
        float blk = (pij - P[i*16+cc] - P[a*16+j] + P[a*16+cc]) *
                    rsqrtf((float)((i - a) * (j - cc)));
        float val = (D[a*16+cc] * (float)k + blk) / (float)(k + 1);
        if (val > best) { best = val; bi = a*16 + cc; }   // first-max = argmax
      }
    inds[(b*16 + k) * 256 + t] = bi;
    __syncthreads();
    D[t] = best;
    __syncthreads();
  }
}

// ------------------------------- backtrack ---------------------------------
__global__ __launch_bounds__(32) void k_backtrack(const int* __restrict__ inds,
                                                  int* __restrict__ s1, int* __restrict__ e1,
                                                  int* __restrict__ s2, int* __restrict__ e2) {
  int b = threadIdx.x;
  if (b >= B_) return;
  int i = T_-1, j = T_-1, ee1 = T_, ee2 = T_;
  for (int k = NSTEP_ - 1; k >= 1; --k) {
    int ind = inds[(b*16 + k) * 256 + i*16 + j];
    int a = ind >> 4, c = ind & 15;
    s1[b*NSTEP_+k] = a + 1; e1[b*NSTEP_+k] = ee1;
    s2[b*NSTEP_+k] = c + 1; e2[b*NSTEP_+k] = ee2;
    i = a; j = c; ee1 = a + 1; ee2 = c + 1;
  }
  s1[b*NSTEP_] = 0; e1[b*NSTEP_] = ee1;
  s2[b*NSTEP_] = 0; e2[b*NSTEP_] = ee2;
}

// ------------- segment linear resize to FIX_ frames + sinusoidal PE --------
__global__ __launch_bounds__(256) void k_seg_pe(const float* __restrict__ f,
                                                const int* __restrict__ ss, const int* __restrict__ ee,
                                                float* __restrict__ h, unsigned short* __restrict__ hbf) {
  int idx = blockIdx.x * blockDim.x + threadIdx.x;     // B*NSTEP*FIX*DIM = 43008
  if (idx >= B_*NSTEP_*FIX_*DIM_) return;
  int d = idx & (DIM_-1); int rest = idx >> 7;
  int fp = rest % FIX_; rest /= FIX_;
  int st = rest % NSTEP_; int b = rest / NSTEP_;
  int s = ss[b*NSTEP_+st], en = ee[b*NSTEP_+st];
  int L = en - s; float Lf = (float)L;
  float src = ((float)fp + 0.5f) * Lf / (float)FIX_ - 0.5f;
  src = fminf(fmaxf(src, 0.f), fmaxf(Lf - 1.f, 0.f));
  int i0 = (int)floorf(src);
  int i1 = min(i0 + 1, L - 1);
  float w = src - (float)i0;
  const float* F = f + (size_t)b * T_ * DIM_;
  float val = F[(s+i0)*DIM_ + d] * (1.f - w) + F[(s+i1)*DIM_ + d] * w;
  int m2 = (d >> 1) * 2;
  float arg = (float)fp * expf(-logf(10000.f) * (float)m2 / (float)DIM_);
  float pe = (d & 1) ? cosf(arg) : sinf(arg);
  float hv = val + pe;
  h[idx] = hv;
  hbf[idx] = f2bf(hv);
}

// -------------------- 3x3 single-head attention per clip -------------------
__global__ __launch_bounds__(128) void k_attn(const float* __restrict__ qkv,
                                              unsigned short* __restrict__ obf) {
  int m = blockIdx.x, d = threadIdx.x;                 // 112 clips, 128 threads
  __shared__ float q[3][DIM_], kk[3][DIM_], v[3][DIM_], s[9];
  for (int i = 0; i < FIX_; ++i) {
    const float* row = qkv + (size_t)(m*FIX_ + i) * (3*DIM_);
    q[i][d] = row[d]; kk[i][d] = row[DIM_ + d]; v[i][d] = row[2*DIM_ + d];
  }
  __syncthreads();
  if (d < 9) {
    int i = d / 3, j = d % 3; float acc = 0.f;
    for (int c = 0; c < DIM_; ++c) acc += q[i][c] * kk[j][c];
    s[d] = acc * rsqrtf((float)DIM_);
  }
  __syncthreads();
  if (d < 3) {
    float mx = fmaxf(fmaxf(s[d*3], s[d*3+1]), s[d*3+2]);
    float e0 = expf(s[d*3]-mx), e1 = expf(s[d*3+1]-mx), e2 = expf(s[d*3+2]-mx);
    float sm = e0 + e1 + e2;
    s[d*3] = e0/sm; s[d*3+1] = e1/sm; s[d*3+2] = e2/sm;
  }
  __syncthreads();
  for (int i = 0; i < FIX_; ++i) {
    float o = s[i*3]*v[0][d] + s[i*3+1]*v[1][d] + s[i*3+2]*v[2][d];
    obf[(size_t)(m*FIX_ + i) * DIM_ + d] = f2bf(o);
  }
}

// ---------------------- residual add + LayerNorm ---------------------------
__global__ __launch_bounds__(128) void k_addln(const float* __restrict__ x,
                                               const float* __restrict__ y,
                                               const float* __restrict__ gs,
                                               const float* __restrict__ gb,
                                               float* __restrict__ out,
                                               unsigned short* __restrict__ outbf) {
  int t = blockIdx.x, d = threadIdx.x;
  __shared__ float red[DIM_];
  __shared__ float mean_s, rstd_s;
  float h = x[(size_t)t*DIM_ + d] + y[(size_t)t*DIM_ + d];
  red[d] = h; __syncthreads();
  for (int s = 64; s > 0; s >>= 1) { if (d < s) red[d] += red[d+s]; __syncthreads(); }
  if (d == 0) mean_s = red[0] / (float)DIM_;
  __syncthreads();
  float c = h - mean_s;
  red[d] = c * c; __syncthreads();
  for (int s = 64; s > 0; s >>= 1) { if (d < s) red[d] += red[d+s]; __syncthreads(); }
  if (d == 0) rstd_s = rsqrtf(red[0] / (float)DIM_ + 1e-5f);
  __syncthreads();
  float o = c * rstd_s * gs[d] + gb[d];
  out[(size_t)t*DIM_ + d] = o;
  if (outbf) outbf[(size_t)t*DIM_ + d] = f2bf(o);
}

// ------------------------- attention pooling -------------------------------
__global__ __launch_bounds__(128) void k_pool(const float* __restrict__ hf,
                                              const float* __restrict__ aw,
                                              const float* __restrict__ ab,
                                              float* __restrict__ out) {
  int m = blockIdx.x, d = threadIdx.x;
  __shared__ float red[DIM_], w[3];
  for (int i = 0; i < FIX_; ++i) {
    red[d] = hf[(size_t)(m*FIX_ + i)*DIM_ + d] * aw[d];
    __syncthreads();
    for (int s = 64; s > 0; s >>= 1) { if (d < s) red[d] += red[d+s]; __syncthreads(); }
    if (d == 0) w[i] = red[0] + ab[0];
    __syncthreads();
  }
  if (d == 0) {
    float mx = fmaxf(fmaxf(w[0], w[1]), w[2]);
    float e0 = expf(w[0]-mx), e1 = expf(w[1]-mx), e2 = expf(w[2]-mx);
    float sm = e0 + e1 + e2;
    w[0] = e0/sm; w[1] = e1/sm; w[2] = e2/sm;
  }
  __syncthreads();
  float o = w[0]*hf[(size_t)(m*FIX_+0)*DIM_+d] +
            w[1]*hf[(size_t)(m*FIX_+1)*DIM_+d] +
            w[2]*hf[(size_t)(m*FIX_+2)*DIM_+d];
  out[(size_t)m*DIM_ + d] = o;
}

// ===========================================================================
// Host orchestration
// ===========================================================================
static void gemm(const unsigned short* A, const unsigned short* Bt, float* C,
                 unsigned short* Cbf, const float* bias, int M, int N, int K,
                 float scale, int relu, hipStream_t s) {
  int tiles = ((M + 15) / 16) * ((N + 15) / 16);
  int blocks = (tiles + 7) / 8;
  k_gemm_bf16<<<blocks, 256, 0, s>>>(A, Bt, C, Cbf, bias, M, N, K, scale, relu);
}
static void cvt(const float* in, unsigned short* out, int n, hipStream_t s) {
  k_f2bf<<<(n + 255) / 256, 256, 0, s>>>(in, out, n);
}

extern "C" void kernel_launch(void* const* d_in, const int* in_sizes, int n_in,
                              void* d_out, int out_size, void* d_ws, size_t ws_size,
                              hipStream_t stream) {
  const float* x1     = (const float*)d_in[0];
  const float* x2     = (const float*)d_in[1];
  const float* bott_w = (const float*)d_in[2];
  const float* bott_b = (const float*)d_in[3];
  const float* se1_w  = (const float*)d_in[4];
  const float* se1_b  = (const float*)d_in[5];
  const float* se2_w  = (const float*)d_in[6];
  const float* se2_b  = (const float*)d_in[7];
  const float* in_w   = (const float*)d_in[8];
  const float* in_b   = (const float*)d_in[9];
  const float* out_w  = (const float*)d_in[10];
  const float* out_b  = (const float*)d_in[11];
  const float* ln1_s  = (const float*)d_in[12];
  const float* ln1_b  = (const float*)d_in[13];
  const float* ff1_w  = (const float*)d_in[14];
  const float* ff1_b  = (const float*)d_in[15];
  const float* ff2_w  = (const float*)d_in[16];
  const float* ff2_b  = (const float*)d_in[17];
  const float* ln2_s  = (const float*)d_in[18];
  const float* ln2_b  = (const float*)d_in[19];
  const float* att_w  = (const float*)d_in[20];
  const float* att_b  = (const float*)d_in[21];
  const float* glob_w = (const float*)d_in[22];
  const float* glob_b = (const float*)d_in[23];

  // Output regions (f32): g1, g2, f1, f2, step1, step2
  float* out  = (float*)d_out;
  float* g1    = out;                 // 1024
  float* g2    = out + 1024;          // 1024
  float* f1    = out + 2048;          // 16384
  float* f2    = out + 18432;         // 16384
  float* step1 = out + 34816;         // 14336
  float* step2 = out + 49152;         // 14336

  // ---- workspace bump allocator ----
  char* base = (char*)d_ws; size_t off = 0;
  auto wsa = [&](size_t bytes) -> void* {
    void* p = base + off; off = (off + bytes + 255) & ~(size_t)255; return p;
  };
  const int KBOT = CIN_ * 9;  // 18432
  unsigned short* Sbf    = (unsigned short*)wsa((size_t)128 * KBOT * 2);
  unsigned short* Wbot   = (unsigned short*)wsa((size_t)DIM_ * KBOT * 2);
  float*          tok    = (float*)wsa(128 * DIM_ * 4);
  unsigned short* Ase1   = (unsigned short*)wsa(128 * DIM_ * 3 * 2);
  unsigned short* se1wbf = (unsigned short*)wsa(DIM_ * DIM_ * 3 * 2);
  float*          z1     = (float*)wsa(128 * DIM_ * 4);
  unsigned short* z1bf   = (unsigned short*)wsa(128 * DIM_ * 2);
  unsigned short* se2wbf = (unsigned short*)wsa(DIM_ * DIM_ * 2);
  unsigned short* f1bf   = (unsigned short*)wsa(B_ * T_ * DIM_ * 2);
  unsigned short* f2bfb  = (unsigned short*)wsa(B_ * T_ * DIM_ * 2);
  float*          pred   = (float*)wsa(B_ * 256 * 4);
  int*            inds   = (int*)wsa(B_ * 16 * 256 * 4);
  int*            s1     = (int*)wsa(B_ * NSTEP_ * 4);
  int*            e1     = (int*)wsa(B_ * NSTEP_ * 4);
  int*            s2     = (int*)wsa(B_ * NSTEP_ * 4);
  int*            e2     = (int*)wsa(B_ * NSTEP_ * 4);
  unsigned short* inwbf  = (unsigned short*)wsa(3 * DIM_ * DIM_ * 2);
  unsigned short* outwbf = (unsigned short*)wsa(DIM_ * DIM_ * 2);
  unsigned short* ff1wbf = (unsigned short*)wsa(DFF_ * DIM_ * 2);
  unsigned short* ff2wbf = (unsigned short*)wsa(DIM_ * DFF_ * 2);
  unsigned short* globwbf= (unsigned short*)wsa(DIM_ * T_ * DIM_ * 2);
  const int TOK = B_ * NSTEP_ * FIX_;   // 336 tokens
  float*          hb     = (float*)wsa(TOK * DIM_ * 4);
  unsigned short* hbbf   = (unsigned short*)wsa(TOK * DIM_ * 2);
  float*          qkv    = (float*)wsa(TOK * 3 * DIM_ * 4);
  unsigned short* obf    = (unsigned short*)wsa(TOK * DIM_ * 2);
  float*          attn   = (float*)wsa(TOK * DIM_ * 4);
  float*          h2     = (float*)wsa(TOK * DIM_ * 4);
  unsigned short* h2bf   = (unsigned short*)wsa(TOK * DIM_ * 2);
  float*          ffr    = (float*)wsa((size_t)TOK * DFF_ * 4);
  unsigned short* ffrbf  = (unsigned short*)wsa((size_t)TOK * DFF_ * 2);
  float*          ffo    = (float*)wsa(TOK * DIM_ * 4);
  float*          hf     = (float*)wsa(TOK * DIM_ * 4);
  (void)ws_size; (void)n_in; (void)in_sizes; (void)out_size;

  // ---- weights -> bf16 (every call; deterministic, graph-safe) ----
  cvt(bott_w, Wbot,    DIM_ * KBOT, stream);
  cvt(se1_w,  se1wbf,  DIM_ * DIM_ * 3, stream);
  cvt(se2_w,  se2wbf,  DIM_ * DIM_, stream);
  cvt(in_w,   inwbf,   3 * DIM_ * DIM_, stream);
  cvt(out_w,  outwbf,  DIM_ * DIM_, stream);
  cvt(ff1_w,  ff1wbf,  DFF_ * DIM_, stream);
  cvt(ff2_w,  ff2wbf,  DIM_ * DFF_, stream);
  cvt(glob_w, globwbf, DIM_ * T_ * DIM_, stream);

  // ---- features: conv+GAP (commuted) -> step encoder -> f ----
  auto features = [&](const float* x, float* fout, unsigned short* fbfout) {
    k_reduce_gap<<<(128 * CIN_) / 256, 256, 0, stream>>>(x, Sbf);
    gemm(Sbf, Wbot, tok, nullptr, bott_b, 128, DIM_, KBOT, 1.f / 49.f, 0, stream);
    k_im2col<<<(128 * DIM_ * 3 + 255) / 256, 256, 0, stream>>>(tok, Ase1);
    gemm(Ase1, se1wbf, z1, z1bf, se1_b, 128, DIM_, DIM_ * 3, 1.f, 1, stream);
    gemm(z1bf, se2wbf, fout, fbfout, se2_b, 128, DIM_, DIM_, 1.f, 0, stream);
  };
  features(x1, f1, f1bf);
  features(x2, f2, f2bfb);

  // ---- similarity + integral image + DP + backtrack ----
  k_sim_pred<<<B_, 256, 0, stream>>>(f1, f2, pred);
  k_dp<<<B_, 256, 0, stream>>>(pred, inds);
  k_backtrack<<<1, 32, 0, stream>>>(inds, s1, e1, s2, e2);

  // ---- temporal transformer + pooling (shared scratch, run twice) ----
  auto temporal = [&](const float* f, const int* ss, const int* ee, float* stepo) {
    k_seg_pe<<<(B_ * NSTEP_ * FIX_ * DIM_ + 255) / 256, 256, 0, stream>>>(f, ss, ee, hb, hbbf);
    gemm(hbbf, inwbf, qkv, nullptr, in_b, TOK, 3 * DIM_, DIM_, 1.f, 0, stream);
    k_attn<<<B_ * NSTEP_, 128, 0, stream>>>(qkv, obf);
    gemm(obf, outwbf, attn, nullptr, out_b, TOK, DIM_, DIM_, 1.f, 0, stream);
    k_addln<<<TOK, 128, 0, stream>>>(hb, attn, ln1_s, ln1_b, h2, h2bf);
    gemm(h2bf, ff1wbf, ffr, ffrbf, ff1_b, TOK, DFF_, DIM_, 1.f, 1, stream);
    gemm(ffrbf, ff2wbf, ffo, nullptr, ff2_b, TOK, DIM_, DFF_, 1.f, 0, stream);
    k_addln<<<TOK, 128, 0, stream>>>(h2, ffo, ln2_s, ln2_b, hf, nullptr);
    k_pool<<<B_ * NSTEP_, 128, 0, stream>>>(hf, att_w, att_b, stepo);
  };
  temporal(f1, s1, e1, step1);
  temporal(f2, s2, e2, step2);

  // ---- global head: (B x T*DIM) @ glob_w^T + b, M=8 handled by guards ----
  gemm(f1bf,  globwbf, g1, nullptr, glob_b, B_, DIM_, T_ * DIM_, 1.f, 0, stream);
  gemm(f2bfb, globwbf, g2, nullptr, glob_b, B_, DIM_, T_ * DIM_, 1.f, 0, stream);
}
